// ConvDualAttention_68736656605716
// MI455X (gfx1250) — matrix operand order repaired and checked by reading, not compiled, not run on hardware
//
#include <hip/hip_runtime.h>

// ---------------------------------------------------------------------------
// ConvDualAttention for MI455X (gfx1250, wave32, WMMA)
// bf16 operands + f32 accumulation via v_wmma_f32_16x16x32_bf16.
// All GEMMs are LDS-free: layouts/orientations chosen so both WMMA operands
// are K-contiguous in global memory -> fragments load as 2x global_load_b128,
// reuse served by the 192MB L2. No barriers in any GEMM kernel.
// ---------------------------------------------------------------------------

#define B_     8
#define DIM_   128
#define HEADS_ 8
#define INNER_ 1024
#define X_     4096
#define NHEAD_ 64           // B_*HEADS_
#define EPS_   1e-5f
#define SCALE_ 0.08838834764831845f   // 128^-0.5

typedef __attribute__((ext_vector_type(16))) __bf16 bf16x16;
typedef __attribute__((ext_vector_type(8)))  __bf16 bf16x8;
typedef __attribute__((ext_vector_type(8)))  float  f32x8;
typedef __attribute__((ext_vector_type(4)))  float  f32x4;

// ---- WMMA helper (CDNA5 16x16x32 bf16, f32 accumulate) --------------------
__device__ __forceinline__ f32x8 wmma_bf16(bf16x16 a, bf16x16 b, f32x8 c) {
  return __builtin_amdgcn_wmma_f32_16x16x32_bf16(false, a, false, b, (short)0, c,
                                                 false, false);
}

// A fragment direct from global: source row-major [M][K] (K contiguous),
// row stride in elements. ISA layout: lane(m,half) elem j<8 <-> K=j+8*half,
// elem j>=8 <-> K=j+8+8*half  ==> two contiguous 8-elem chunks.
__device__ __forceinline__ bf16x16 loadA_direct(const __bf16* base,
                                                size_t row_stride, int m0,
                                                int k0) {
  const int lane = threadIdx.x & 31;
  const int half = lane >> 4;
  const __bf16* r = base + (size_t)(m0 + (lane & 15)) * row_stride + k0;
  bf16x8 lo = *(const bf16x8*)(r + 8 * half);
  bf16x8 hi = *(const bf16x8*)(r + 16 + 8 * half);
  bf16x16 a;
#pragma unroll
  for (int i = 0; i < 8; ++i) { a[i] = lo[i]; a[8 + i] = hi[i]; }
  return a;
}

// B fragment direct from global: source rows indexed by N, K contiguous
// ([N][K] layout). ISA layout: lane(n,half) elem j<8 <-> K=j+8*half,
// elem j>=8 <-> K=j+8+8*half (same as A).
__device__ __forceinline__ bf16x16 loadB_direct(const __bf16* base,
                                                size_t row_stride, int n0,
                                                int k0) {
  const int lane = threadIdx.x & 31;
  const int half = lane >> 4;
  const __bf16* r =
      base + (size_t)(n0 + (lane & 15)) * row_stride + k0;
  bf16x8 lo = *(const bf16x8*)(r + 8 * half);
  bf16x8 hi = *(const bf16x8*)(r + 16 + 8 * half);
  bf16x16 b;
#pragma unroll
  for (int i = 0; i < 8; ++i) { b[i] = lo[i]; b[8 + i] = hi[i]; }
  return b;
}

// ---- 1. BN folding into pointwise weights ---------------------------------
__global__ __launch_bounds__(DIM_) void fold_w_kernel(
    const float* __restrict__ pw, const float* __restrict__ g,
    const float* __restrict__ bb, const float* __restrict__ mm,
    const float* __restrict__ vv, const float* __restrict__ dw,
    __bf16* __restrict__ W, float* __restrict__ bias) {
  const int o = blockIdx.x;
  const int c = threadIdx.x;
  const float s  = g[c] * rsqrtf(vv[c] + EPS_);
  const float sc = s * (dw ? dw[c] : 1.f);
  const float w  = pw[o * DIM_ + c];
  W[o * DIM_ + c] = (__bf16)(w * sc);
  __shared__ float red[DIM_];
  red[c] = w * (bb[c] - mm[c] * s);
  __syncthreads();
  for (int st = DIM_ / 2; st > 0; st >>= 1) {
    if (c < st) red[c] += red[c + st];
    __syncthreads();
  }
  if (c == 0) bias[o] = red[0];
}

__global__ void cvt_bf16_kernel(const float* __restrict__ src,
                                __bf16* __restrict__ dst, int n) {
  const int i = blockIdx.x * 256 + threadIdx.x;
  if (i < n) dst[i] = (__bf16)src[i];
}

// ---- 2. depthwise conv3 (pad 1) + transpose: Yt[p][b][t][c] (c contiguous) -
__global__ __launch_bounds__(256) void dwconv_t_kernel(
    const float* __restrict__ x, const float* __restrict__ qdw,
    const float* __restrict__ kdw, const float* __restrict__ vdw,
    __bf16* __restrict__ Yt) {
  __shared__ __align__(32) __bf16 tile[64][136];   // [t][c], 272B rows (16B mult)
  const int t0 = blockIdx.x * 64;
  const int b  = blockIdx.y;
  const int p  = blockIdx.z;
  const float* dw = (p == 0 ? qdw : (p == 1 ? kdw : vdw));
  for (int idx = threadIdx.x; idx < DIM_ * 64; idx += 256) {
    const int c = idx >> 6, tt = idx & 63;
    const int t = t0 + tt;
    const float* xr = x + ((size_t)b * DIM_ + c) * X_;
    const float xm = (t > 0)      ? xr[t - 1] : 0.f;
    const float xc = xr[t];
    const float xp = (t < X_ - 1) ? xr[t + 1] : 0.f;
    tile[tt][c] = (__bf16)(dw[c * 3] * xm + dw[c * 3 + 1] * xc + dw[c * 3 + 2] * xp);
  }
  __syncthreads();
  for (int idx = threadIdx.x; idx < 64 * 16; idx += 256) {  // b128 writes
    const int tt = idx >> 4, c8 = (idx & 15) * 8;
    bf16x8 v = *(const bf16x8*)&tile[tt][c8];
    *(bf16x8*)(Yt + (((size_t)p * B_ + b) * X_ + t0 + tt) * DIM_ + c8) = v;
  }
}

// ---- 3a. projection GEMM, M=t N=o  ->  Pdt[n][d][t] (t contiguous) --------
__global__ __launch_bounds__(128) void proj_dt_kernel(
    const __bf16* __restrict__ W,    // [INNER_][DIM_]
    const __bf16* __restrict__ Yt,   // [B_][X_][DIM_] (this projection)
    const float* __restrict__ bias, __bf16* __restrict__ Pdt) {
  const int wave = threadIdx.x >> 5, lane = threadIdx.x & 31;
  const int o0 = blockIdx.x * 64 + wave * 16;   // N (per wave)
  const int t0 = blockIdx.y * 16;               // M
  const int b  = blockIdx.z;
  const __bf16* Yb = Yt + (size_t)b * X_ * DIM_;
  f32x8 acc = {};
#pragma unroll
  for (int k0 = 0; k0 < DIM_; k0 += 32)
    acc = wmma_bf16(loadA_direct(Yb, DIM_, t0, k0),
                    loadB_direct(W, DIM_, o0, k0), acc);
  const int o    = o0 + (lane & 15);
  const int half = lane >> 4;
  const int nh   = (b << 3) + (o >> 7);
  const int d    = o & 127;
  const float bi = bias[o];
  bf16x8 ov;
#pragma unroll
  for (int r = 0; r < 8; ++r) ov[r] = (__bf16)(acc[r] + bi);
  *(bf16x8*)(Pdt + ((size_t)nh * DIM_ + d) * X_ + t0 + 8 * half) = ov;
}

// ---- 3b. projection GEMM, M=o N=t  ->  Ptd[n][t][d] (d contiguous) --------
__global__ __launch_bounds__(128) void proj_td_kernel(
    const __bf16* __restrict__ W, const __bf16* __restrict__ Yt,
    const float* __restrict__ bias, __bf16* __restrict__ Ptd) {
  const int wave = threadIdx.x >> 5, lane = threadIdx.x & 31;
  const int t0 = blockIdx.x * 64 + wave * 16;   // N (per wave)
  const int o0 = blockIdx.y * 16;               // M
  const int b  = blockIdx.z;
  const __bf16* Yb = Yt + (size_t)b * X_ * DIM_;
  f32x8 acc = {};
#pragma unroll
  for (int k0 = 0; k0 < DIM_; k0 += 32)
    acc = wmma_bf16(loadA_direct(W, DIM_, o0, k0),
                    loadB_direct(Yb, DIM_, t0, k0), acc);
  const int t       = t0 + (lane & 15);
  const int half    = lane >> 4;
  const int o_first = o0 + 8 * half;
  const int nh      = (b << 3) + (o_first >> 7);
  const int d0      = o_first & 127;
  bf16x8 ov;
#pragma unroll
  for (int r = 0; r < 8; ++r) ov[r] = (__bf16)(acc[r] + bias[o_first + r]);
  *(bf16x8*)(Ptd + ((size_t)nh * X_ + t) * DIM_ + d0) = ov;
}

// ---- 4. softmax over d on Kt[n][d][t] (thread-per-t, coalesced) -----------
__global__ __launch_bounds__(256) void softmax_dt_kernel(__bf16* __restrict__ K) {
  const int t = blockIdx.x * 256 + threadIdx.x;
  const int n = blockIdx.y;
  __bf16* base = K + (size_t)n * DIM_ * X_ + t;
  float mx = -3.0e38f;
  for (int d = 0; d < DIM_; ++d) mx = fmaxf(mx, (float)base[(size_t)d * X_]);
  float sum = 0.f;
  for (int d = 0; d < DIM_; ++d) sum += __expf((float)base[(size_t)d * X_] - mx);
  const float inv = 1.f / sum;
  for (int d = 0; d < DIM_; ++d) {
    const float e = __expf((float)base[(size_t)d * X_] - mx);
    base[(size_t)d * X_] = (__bf16)(e * inv);
  }
}

// ---- 5. kat[n][e][d] = SCALE * sum_t q[d][t] * sk[e][t]  (M=d, N=e) -------
__global__ __launch_bounds__(128) void kat_gemm_kernel(
    const __bf16* __restrict__ Qd,   // [n][d][t]
    const __bf16* __restrict__ Kt,   // [n][e][t] (softmaxed)
    __bf16* __restrict__ KAT) {      // [n][e][d]
  const int wave = threadIdx.x >> 5, lane = threadIdx.x & 31;
  const int e0 = blockIdx.x * 64 + wave * 16;   // N (per wave)
  const int d0 = blockIdx.y * 16;               // M
  const int n  = blockIdx.z;
  const __bf16* q  = Qd + (size_t)n * DIM_ * X_;
  const __bf16* sk = Kt + (size_t)n * DIM_ * X_;
  f32x8 acc = {};
  for (int k0 = 0; k0 < X_; k0 += 32) {
    if (k0 + 32 < X_) {
      __builtin_prefetch(q  + (size_t)(d0 + (lane & 15)) * X_ + k0 + 32, 0, 1);
      __builtin_prefetch(sk + (size_t)(e0 + (lane & 15)) * X_ + k0 + 32, 0, 1);
    }
    acc = wmma_bf16(loadA_direct(q, X_, d0, k0),
                    loadB_direct(sk, X_, e0, k0), acc);
  }
  const int e    = e0 + (lane & 15);
  const int half = lane >> 4;
  bf16x8 ov;
#pragma unroll
  for (int r = 0; r < 8; ++r) ov[r] = (__bf16)(acc[r] * SCALE_);
  *(bf16x8*)(KAT + ((size_t)n * DIM_ + e) * DIM_ + d0 + 8 * half) = ov;
}

// ---- 6. out[n][t][e] = (v@kat)[t][e] + sigmoid(q@Wg^T+bg)[t][e]*v[t][e] ----
// Orientation M=e, N=t: A1=KAT[e][d], B1=Vr[t][d], A2=WG[e][i], B2=Qr[t][i]
__global__ __launch_bounds__(128) void gate_out_kernel(
    const __bf16* __restrict__ Qr,   // [n][t][d]
    const __bf16* __restrict__ Vr,   // [n][t][d]
    const __bf16* __restrict__ KAT,  // [n][e][d]
    const __bf16* __restrict__ WG,   // [DIM_][DIM_]
    const float* __restrict__ BG, __bf16* __restrict__ OUT /* [n][t][e] */) {
  const int wave = threadIdx.x >> 5, lane = threadIdx.x & 31;
  const int t0 = blockIdx.x * 64 + wave * 16;   // N (per wave)
  const int e0 = blockIdx.y * 16;               // M
  const int n  = blockIdx.z;
  const __bf16* q   = Qr  + (size_t)n * X_ * DIM_;
  const __bf16* v   = Vr  + (size_t)n * X_ * DIM_;
  const __bf16* kat = KAT + (size_t)n * DIM_ * DIM_;
  f32x8 accV = {};
  f32x8 accG = {};
#pragma unroll
  for (int k0 = 0; k0 < DIM_; k0 += 32) {
    accV = wmma_bf16(loadA_direct(kat, DIM_, e0, k0),
                     loadB_direct(v, DIM_, t0, k0), accV);
    accG = wmma_bf16(loadA_direct(WG, DIM_, e0, k0),
                     loadB_direct(q, DIM_, t0, k0), accG);
  }
  const int t      = t0 + (lane & 15);
  const int half   = lane >> 4;
  const int efirst = e0 + 8 * half;
  bf16x8 vv = *(const bf16x8*)(v + (size_t)t * DIM_ + efirst);
  bf16x8 ov;
#pragma unroll
  for (int r = 0; r < 8; ++r) {
    const float g    = accG[r] + BG[efirst + r];
    const float gate = (float)vv[r] / (1.f + __expf(-g));
    ov[r] = (__bf16)(accV[r] + gate);
  }
  *(bf16x8*)(OUT + ((size_t)n * X_ + t) * DIM_ + efirst) = ov;
}

// ---- 7. final[b][o][t] = sum_i Wout[o][i]*OUT[b*8+i/128][t][i%128] + out_b --
// Orientation M=t, N=o: A=OUT per-head [t][e], B=WOUT[o][i]
__global__ __launch_bounds__(128) void final_gemm_kernel(
    const __bf16* __restrict__ WOUT, // [DIM_][INNER_]
    const __bf16* __restrict__ OUT,  // [NHEAD_][X_][DIM_]
    const float* __restrict__ outb, float* __restrict__ dst) {
  const int wave = threadIdx.x >> 5, lane = threadIdx.x & 31;
  const int o0 = blockIdx.x * 64 + wave * 16;   // N (per wave)
  const int t0 = blockIdx.y * 16;               // M
  const int b  = blockIdx.z;
  f32x8 acc = {};
  for (int k0 = 0; k0 < INNER_; k0 += 32) {
    // 32-wide K block stays within one head (128 % 32 == 0)
    const __bf16* abase =
        OUT + (size_t)(b * 8 + (k0 >> 7)) * X_ * DIM_ + (k0 & 127);
    if (k0 + 32 < INNER_)
      __builtin_prefetch(OUT + (size_t)(b * 8 + ((k0 + 32) >> 7)) * X_ * DIM_ +
                             (size_t)(t0 + (lane & 15)) * DIM_ + ((k0 + 32) & 127),
                         0, 1);
    acc = wmma_bf16(loadA_direct(abase, DIM_, t0, 0),
                    loadB_direct(WOUT, INNER_, o0, k0), acc);
  }
  const int o    = o0 + (lane & 15);
  const int half = lane >> 4;
  const float bi = outb[o];
  float* dp = dst + ((size_t)b * DIM_ + o) * X_ + t0 + 8 * half;
  f32x4 lo, hi;
#pragma unroll
  for (int r = 0; r < 4; ++r) { lo[r] = acc[r] + bi; hi[r] = acc[r + 4] + bi; }
  *(f32x4*)(dp)     = lo;
  *(f32x4*)(dp + 4) = hi;
}

// ---------------------------------------------------------------------------

extern "C" void kernel_launch(void* const* d_in, const int* in_sizes, int n_in,
                              void* d_out, int out_size, void* d_ws,
                              size_t ws_size, hipStream_t stream) {
  (void)in_sizes; (void)n_in; (void)out_size; (void)ws_size;
  const float* x     = (const float*)d_in[0];
  const float* q_dw  = (const float*)d_in[1];
  const float* q_g   = (const float*)d_in[2];
  const float* q_b   = (const float*)d_in[3];
  const float* q_m   = (const float*)d_in[4];
  const float* q_v   = (const float*)d_in[5];
  const float* q_pw  = (const float*)d_in[6];
  const float* k_dw  = (const float*)d_in[7];
  const float* k_g   = (const float*)d_in[8];
  const float* k_b   = (const float*)d_in[9];
  const float* k_m   = (const float*)d_in[10];
  const float* k_v   = (const float*)d_in[11];
  const float* k_pw  = (const float*)d_in[12];
  const float* v_dw  = (const float*)d_in[13];
  const float* v_g   = (const float*)d_in[14];
  const float* v_b   = (const float*)d_in[15];
  const float* v_m   = (const float*)d_in[16];
  const float* v_v   = (const float*)d_in[17];
  const float* v_pw  = (const float*)d_in[18];
  const float* gt_dw = (const float*)d_in[19];
  const float* gt_g  = (const float*)d_in[20];
  const float* gt_b  = (const float*)d_in[21];
  const float* gt_m  = (const float*)d_in[22];
  const float* gt_v  = (const float*)d_in[23];
  const float* gt_pw = (const float*)d_in[24];
  const float* out_w = (const float*)d_in[25];
  const float* out_b = (const float*)d_in[26];

  // --- workspace carve-out (deterministic) ---
  char*  ws  = (char*)d_ws;
  size_t off = 0;
  auto alloc = [&](size_t bytes) -> void* {
    void* p = ws + off;
    off = (off + bytes + 255) & ~(size_t)255;
    return p;
  };
  const size_t YSZ = (size_t)B_ * X_ * DIM_;        // per projection
  const size_t PSZ = (size_t)NHEAD_ * X_ * DIM_;    // q/k/v/out
  __bf16* WQ   = (__bf16*)alloc(INNER_ * DIM_ * 2);
  __bf16* WK   = (__bf16*)alloc(INNER_ * DIM_ * 2);
  __bf16* WV   = (__bf16*)alloc(INNER_ * DIM_ * 2);
  __bf16* WG   = (__bf16*)alloc(DIM_ * DIM_ * 2);
  __bf16* WOUT = (__bf16*)alloc(DIM_ * INNER_ * 2);
  float*  BQ   = (float*)alloc(INNER_ * 4);
  float*  BK   = (float*)alloc(INNER_ * 4);
  float*  BV   = (float*)alloc(INNER_ * 4);
  float*  BG   = (float*)alloc(DIM_ * 4);
  __bf16* Yt   = (__bf16*)alloc(3 * YSZ * 2);       // [p][b][t][c]
  __bf16* Qd   = (__bf16*)alloc(PSZ * 2);           // [n][d][t]
  __bf16* Qr   = (__bf16*)alloc(PSZ * 2);           // [n][t][d]
  __bf16* Kt   = (__bf16*)alloc(PSZ * 2);           // [n][d][t]
  __bf16* Vr   = (__bf16*)alloc(PSZ * 2);           // [n][t][d]
  __bf16* KATb = (__bf16*)alloc((size_t)NHEAD_ * DIM_ * DIM_ * 2); // [n][e][d]
  __bf16* OUTb = (__bf16*)alloc(PSZ * 2);           // [n][t][e]

  // 1. fold BN into pointwise weights
  fold_w_kernel<<<INNER_, DIM_, 0, stream>>>(q_pw, q_g, q_b, q_m, q_v, nullptr, WQ, BQ);
  fold_w_kernel<<<INNER_, DIM_, 0, stream>>>(k_pw, k_g, k_b, k_m, k_v, nullptr, WK, BK);
  fold_w_kernel<<<INNER_, DIM_, 0, stream>>>(v_pw, v_g, v_b, v_m, v_v, nullptr, WV, BV);
  fold_w_kernel<<<DIM_, DIM_, 0, stream>>>(gt_pw, gt_g, gt_b, gt_m, gt_v, gt_dw, WG, BG);
  cvt_bf16_kernel<<<(DIM_ * INNER_ + 255) / 256, 256, 0, stream>>>(out_w, WOUT,
                                                                   DIM_ * INNER_);
  // 2. depthwise conv3 + transpose -> Yt[p][b][t][c]
  dwconv_t_kernel<<<dim3(X_ / 64, B_, 3), 256, 0, stream>>>(x, q_dw, k_dw, v_dw, Yt);

  // 3. projections (all LDS-free WMMA GEMMs)
  const dim3 gdt(INNER_ / 64, X_ / 16, B_);
  const dim3 gtd(X_ / 64, INNER_ / 16, B_);
  proj_dt_kernel<<<gdt, 128, 0, stream>>>(WK, Yt + YSZ,     BK, Kt);  // k  -> [n][d][t]
  proj_dt_kernel<<<gdt, 128, 0, stream>>>(WQ, Yt,           BQ, Qd);  // q  -> [n][d][t]
  proj_td_kernel<<<gtd, 128, 0, stream>>>(WQ, Yt,           BQ, Qr);  // q  -> [n][t][d]
  proj_td_kernel<<<gtd, 128, 0, stream>>>(WV, Yt + 2 * YSZ, BV, Vr);  // v  -> [n][t][d]

  // 4. softmax over d on Kt (coalesced over t)
  softmax_dt_kernel<<<dim3(X_ / 256, NHEAD_), 256, 0, stream>>>(Kt);

  // 5. kat = q^T @ softmax(k) * scale  -> [n][e][d]
  kat_gemm_kernel<<<dim3(DIM_ / 64, DIM_ / 16, NHEAD_), 128, 0, stream>>>(Qd, Kt,
                                                                          KATb);
  // 6. out = v @ kat + sigmoid(gate) * v  -> [n][t][e]
  gate_out_kernel<<<dim3(X_ / 64, DIM_ / 16, NHEAD_), 128, 0, stream>>>(
      Qr, Vr, KATb, WG, BG, OUTb);

  // 7. final 1x1 conv (1024 -> 128) + bias -> f32 output
  final_gemm_kernel<<<dim3(DIM_ / 64, X_ / 16, B_), 128, 0, stream>>>(
      WOUT, OUTb, out_b, (float*)d_out);
}
